// ReactomeGNN_27960237097487
// MI455X (gfx1250) — matrix operand order repaired and checked by reading, not compiled, not run on hardware
//
#include <hip/hip_runtime.h>
#include <hip/hip_bf16.h>

typedef __attribute__((ext_vector_type(16))) _Float16 v16h;
typedef __attribute__((ext_vector_type(8)))  _Float16 v8h;
typedef __attribute__((ext_vector_type(8)))  float    v8f;

#define RB   64
#define RM   3
#define RP   32
#define RH   64
#define RTOTAL 12302
#define RNGENES 9229
#define RN   (RB * RTOTAL)          // 787328 nodes
#define RTOP 26
#define REPS 1e-5f

// ---------------------------------------------------------------------------
// Initial node features, stored f16:  h16[N,32]
// ---------------------------------------------------------------------------
__global__ void init_h_kernel(const float* __restrict__ x,
                              const float* __restrict__ Wf,
                              const float* __restrict__ bf,
                              const float* __restrict__ pe,
                              _Float16* __restrict__ h16) {
    long long idx = (long long)blockIdx.x * blockDim.x + threadIdx.x;   // over N*32
    int p = (int)(idx & 31);
    long long n = idx >> 5;
    int b = (int)(n / RTOTAL);
    int i = (int)(n % RTOTAL);
    float v;
    if (i < RNGENES) {
        const float* xr = x + (size_t)b * (RNGENES * RM) + (size_t)i * RM;
        v = xr[0] * Wf[0 * RP + p] + xr[1] * Wf[1 * RP + p] + xr[2] * Wf[2 * RP + p] + bf[p];
    } else {
        v = pe[(size_t)(i - RNGENES) * RP + p];
    }
    h16[(size_t)n * RP + p] = (_Float16)v;
}

// ---------------------------------------------------------------------------
// Degree / norm precompute (buffer transformed in place into d^-1/2)
// ---------------------------------------------------------------------------
__global__ void deg_init_kernel(float* __restrict__ deg) {
    int i = blockIdx.x * blockDim.x + threadIdx.x;
    if (i < RN) deg[i] = 1.0f;                      // self-loop
}
__global__ void deg_edges_kernel(const int* __restrict__ dst, float* __restrict__ deg, int E) {
    int e = blockIdx.x * blockDim.x + threadIdx.x;
    if (e < E) atomicAdd(&deg[dst[e]], 1.0f);
}
__global__ void deg_finalize_kernel(float* __restrict__ deg) {
    int i = blockIdx.x * blockDim.x + threadIdx.x;
    if (i < RN) deg[i] = rsqrtf(deg[i]);            // now holds d^{-1/2}
}

// ---------------------------------------------------------------------------
// Pack W[din,64] (f32) into per-lane WMMA B-fragments (f16):
//   bfrag[kci*2048 + nt*512 + lane*16 + j] = W[(kci*32 + 8*hi + (j&7) + 16*(j>>3))*64 + nt*16 + n]
//   (hi = lane>>4, n = lane&15)
// ---------------------------------------------------------------------------
__global__ void prep_bfrag_kernel(const float* __restrict__ W,
                                  _Float16* __restrict__ bfrag, int nkc) {
    const int total = nkc * 2048;
    for (int t = threadIdx.x; t < total; t += 256) {
        int j    = t & 15;
        int lane = (t >> 4) & 31;
        int nt   = (t >> 9) & 3;
        int kci  = t >> 11;
        int hi   = lane >> 4;
        int n    = lane & 15;
        int K    = kci * 32 + 8 * hi + (j & 7) + ((j >> 3) << 4);
        bfrag[t] = (_Float16)W[(size_t)K * RH + nt * 16 + n];
    }
}

// ---------------------------------------------------------------------------
// WMMA GEMM:  hw16[N,64] = f16( h16[N,din] @ W )   (f32 accumulate)
// Fused epilogue: agg = hw * (1/deg) + bias
// Block = 256 thr = 8 waves; each wave -> 16 rows x 64 cols; exact tiling.
// ---------------------------------------------------------------------------
__global__ __launch_bounds__(256)
void gemm_wmma_kernel(const _Float16* __restrict__ h16,
                      const _Float16* __restrict__ bfrag,   // pre-packed B frags
                      const float* __restrict__ bias,       // [64]
                      const float* __restrict__ dinv,       // [N]
                      _Float16* __restrict__ hw16,
                      float* __restrict__ agg,
                      int din) {
    const int tid  = threadIdx.x;
    const int wave = tid >> 5;
    const int lane = tid & 31;
    const int m    = lane & 15;          // row-in-tile (A) / col (B,C,D)
    const int hi   = lane >> 4;          // lane-group select
    const int m0   = blockIdx.x * 128 + wave * 16;

    // prefetch next row-slab's A data (stays inside workspace allocation)
    __builtin_prefetch(h16 + (size_t)(m0 + 128) * din, 0, 1);

    v8f acc[4];
#pragma unroll
    for (int nt = 0; nt < 4; ++nt)
#pragma unroll
        for (int r = 0; r < 8; ++r) acc[nt][r] = 0.0f;

    const int nkc = din >> 5;            // 1 (layer0) or 2
    for (int kci = 0; kci < nkc; ++kci) {
        const int kc = kci << 5;
        // ---- A fragment: half j -> K = (j<8?j:j+8) + 8*hi ; two b128 loads
        const _Float16* ap = h16 + (size_t)(m0 + m) * din + kc + 8 * hi;
        v8h lo  = *(const v8h*)ap;          // K = kc + 8*hi + [0..7]
        v8h hi8 = *(const v8h*)(ap + 16);   // K = kc + 16 + 8*hi + [0..7]
        v16h a = __builtin_shufflevector(lo, hi8,
                                         0, 1, 2, 3, 4, 5, 6, 7,
                                         8, 9, 10, 11, 12, 13, 14, 15);
#pragma unroll
        for (int nt = 0; nt < 4; ++nt) {
            // ---- B fragment: one aligned 32-B per-lane load
            v16h bfr = *(const v16h*)(bfrag + (size_t)kci * 2048 + nt * 512 + lane * 16);
            acc[nt] = __builtin_amdgcn_wmma_f32_16x16x32_f16(
                false, a, false, bfr, (short)0, acc[nt], false, false);
        }
    }

    // ---- epilogue: VGPR r holds row (r + 8*hi), col = nt*16 + m
#pragma unroll
    for (int r = 0; r < 8; ++r) {
        const int row = m0 + r + 8 * hi;
        const float dn = dinv[row];
        const float sn = dn * dn;                 // 1/deg
#pragma unroll
        for (int nt = 0; nt < 4; ++nt) {
            const int col = nt * 16 + m;
            const float v = acc[nt][r];
            const size_t o = (size_t)row * RH + col;
            hw16[o] = (_Float16)v;
            agg[o]  = v * sn + bias[col];
        }
    }
}

// ---------------------------------------------------------------------------
// Edge scatter: agg[dst] += hw[src] * dinv[src]*dinv[dst]
// 64 contiguous channels per edge -> coalesced f32 atomics
// ---------------------------------------------------------------------------
__global__ void scatter_kernel(const int* __restrict__ src,
                               const int* __restrict__ dst,
                               const float* __restrict__ dinv,
                               const _Float16* __restrict__ hw16,
                               float* __restrict__ agg,
                               int E) {
    long long idx = (long long)blockIdx.x * blockDim.x + threadIdx.x;
    int e = (int)(idx >> 6);
    int c = (int)(idx & 63);
    if (e < E) {
        int s = src[e], d = dst[e];
        float nrm = dinv[s] * dinv[d];
        float hv = (float)hw16[(size_t)s * RH + c];
        atomicAdd(&agg[(size_t)d * RH + c], hv * nrm);
    }
}

// ---------------------------------------------------------------------------
// BatchNorm statistics (sum, sumsq) per channel over all N rows
// ---------------------------------------------------------------------------
__global__ void zero_sums_kernel(float* __restrict__ sums) {
    sums[threadIdx.x] = 0.0f;                     // 128 threads
}
__global__ void bn_stats_kernel(const float* __restrict__ agg,
                                float* __restrict__ sums) {
    int tid = threadIdx.x;
    int c = tid & 63;
    int g = tid >> 6;                             // 0..3 row-groups per block
    float s = 0.0f, s2 = 0.0f;
    for (int r = blockIdx.x * 4 + g; r < RN; r += gridDim.x * 4) {
        float v = agg[(size_t)r * RH + c];
        s += v; s2 += v * v;
    }
    __shared__ float ls[256], ls2[256];
    ls[tid] = s; ls2[tid] = s2;
    __syncthreads();
    if (g == 0) {
        s  = ls[c]  + ls[c + 64]  + ls[c + 128]  + ls[c + 192];
        s2 = ls2[c] + ls2[c + 64] + ls2[c + 128] + ls2[c + 192];
        atomicAdd(&sums[c], s);
        atomicAdd(&sums[64 + c], s2);
    }
}

// ---------------------------------------------------------------------------
// BatchNorm apply + tanh -> f16 activations
// ---------------------------------------------------------------------------
__global__ void bn_apply_kernel(const float* __restrict__ agg,
                                const float* __restrict__ sums,
                                const float* __restrict__ gamma,
                                const float* __restrict__ beta,
                                _Float16* __restrict__ h16) {
    long long idx = (long long)blockIdx.x * blockDim.x + threadIdx.x;  // N*64
    int c = (int)(idx & 63);
    const float invN = 1.0f / (float)RN;
    float mu  = sums[c] * invN;
    float var = sums[64 + c] * invN - mu * mu;
    float v = (agg[idx] - mu) * rsqrtf(var + REPS) * gamma[c] + beta[c];
    h16[idx] = (_Float16)tanhf(v);
}

// ---------------------------------------------------------------------------
// Readout: mean over top-26 nodes -> dot Wc -> + bc
// ---------------------------------------------------------------------------
__global__ void readout_kernel(const _Float16* __restrict__ h16,
                               const float* __restrict__ Wc,
                               const float* __restrict__ bc,
                               float* __restrict__ out) {
    int b = blockIdx.x;
    int c = threadIdx.x;                          // 64 threads
    size_t base = ((size_t)b * RTOTAL + (RTOTAL - RTOP)) * RH;
    float s = 0.0f;
#pragma unroll
    for (int k = 0; k < RTOP; ++k) s += (float)h16[base + (size_t)k * RH + c];
    s = s * (1.0f / (float)RTOP) * Wc[c];
    __shared__ float red[64];
    red[c] = s;
    __syncthreads();
    for (int off = 32; off > 0; off >>= 1) {
        if (c < off) red[c] += red[c + off];
        __syncthreads();
    }
    if (c == 0) out[b] = red[0] + bc[0];
}

// ---------------------------------------------------------------------------
extern "C" void kernel_launch(void* const* d_in, const int* in_sizes, int n_in,
                              void* d_out, int out_size, void* d_ws, size_t ws_size,
                              hipStream_t stream) {
    const float* x  = (const float*)d_in[0];
    const float* Wf = (const float*)d_in[1];
    const float* bf = (const float*)d_in[2];
    const float* pe = (const float*)d_in[3];

    const float *Wl[6], *bl[6], *gl[6], *betal[6];
    const float *Wc, *bc;
    const int *src, *dst;
    int E;

    if (n_in >= 30) {
        // pytree-leaf layout: each tuple element is its own input
        for (int i = 0; i < 6; ++i) {
            Wl[i]    = (const float*)d_in[4 + i];
            bl[i]    = (const float*)d_in[10 + i];
            gl[i]    = (const float*)d_in[16 + i];
            betal[i] = (const float*)d_in[22 + i];
        }
        Wc  = (const float*)d_in[28];
        bc  = (const float*)d_in[29];
        src = (const int*)d_in[30];
        dst = (const int*)d_in[31];
        E   = in_sizes[30];
    } else {
        // concatenated-tuple layout
        const float* Wall = (const float*)d_in[4];
        const size_t woff[6] = {0, 2048, 6144, 10240, 14336, 18432};
        for (int i = 0; i < 6; ++i) {
            Wl[i]    = Wall + woff[i];
            bl[i]    = (const float*)d_in[5] + i * RH;
            gl[i]    = (const float*)d_in[6] + i * RH;
            betal[i] = (const float*)d_in[7] + i * RH;
        }
        Wc  = (const float*)d_in[8];
        bc  = (const float*)d_in[9];
        src = (const int*)d_in[10];
        dst = (const int*)d_in[11];
        E   = in_sizes[10];
    }

    // workspace layout
    char* ws = (char*)d_ws;
    const size_t S16 = (size_t)RN * RH * sizeof(_Float16);   // 100.8 MB
    const size_t S32 = (size_t)RN * RH * sizeof(float);      // 201.6 MB
    _Float16* h16   = (_Float16*)(ws);
    _Float16* hw16  = (_Float16*)(ws + S16);
    float*    aggbuf = (float*)(ws + 2 * S16);
    float*    dinvb  = (float*)(ws + 2 * S16 + S32);                       // N floats
    _Float16* bfrag  = (_Float16*)(ws + 2 * S16 + S32 + (size_t)RN * 4);   // 4096 halves
    float*    sums   = (float*)(ws + 2 * S16 + S32 + (size_t)RN * 4 + 8192);

    // ---- initial features
    init_h_kernel<<<(unsigned)((size_t)RN * RP / 256), 256, 0, stream>>>(x, Wf, bf, pe, h16);

    // ---- degree / norm
    deg_init_kernel<<<(RN + 255) / 256, 256, 0, stream>>>(dinvb);
    deg_edges_kernel<<<(E + 255) / 256, 256, 0, stream>>>(dst, dinvb, E);
    deg_finalize_kernel<<<(RN + 255) / 256, 256, 0, stream>>>(dinvb);

    // ---- 6 GCN layers
    const unsigned gemm_grid    = RN / 128;                               // 6151, exact
    const unsigned scatter_grid = (unsigned)(((size_t)E * RH + 255) / 256);
    const unsigned apply_grid   = (unsigned)((size_t)RN * RH / 256);      // exact

    for (int l = 0; l < 6; ++l) {
        const int din = (l == 0) ? RP : RH;
        prep_bfrag_kernel<<<1, 256, 0, stream>>>(Wl[l], bfrag, din >> 5);
        gemm_wmma_kernel<<<gemm_grid, 256, 0, stream>>>(h16, bfrag, bl[l], dinvb,
                                                        hw16, aggbuf, din);
        scatter_kernel<<<scatter_grid, 256, 0, stream>>>(src, dst, dinvb, hw16, aggbuf, E);
        zero_sums_kernel<<<1, 128, 0, stream>>>(sums);
        bn_stats_kernel<<<1024, 256, 0, stream>>>(aggbuf, sums);
        bn_apply_kernel<<<apply_grid, 256, 0, stream>>>(aggbuf, sums, gl[l], betal[l], h16);
    }

    // ---- readout
    readout_kernel<<<RB, 64, 0, stream>>>(h16, Wc, bc, (float*)d_out);
}
